// VanillaMHA_59399397703708
// MI455X (gfx1250) — matrix operand order repaired and checked by reading, not compiled, not run on hardware
//
#include <hip/hip_runtime.h>
#include <hip/hip_bf16.h>
#include <math.h>

// Problem constants (from reference): B=2, L=2048, D=1024, H=16, DH=64
constexpr int kB  = 2;
constexpr int kL  = 2048;
constexpr int kD  = 1024;
constexpr int kH  = 16;
constexpr int kDH = 64;
constexpr int kBH = kB * kH;           // 32
constexpr int kM  = kB * kL;           // 4096 rows for the projections

typedef __attribute__((ext_vector_type(16))) _Float16 v16h;
typedef __attribute__((ext_vector_type(8)))  _Float16 v8h;
typedef __attribute__((ext_vector_type(8)))  float    v8f;

// ---------------------------------------------------------------------------
// WMMA fragment loaders, wave32, using contiguous k order for A and B (not the ISA table order).
//   A (16x32 f16): lane (l&15) holds row M=l&15.
//     lanes 0-15 : K = k0+0..15 contiguous in elems 0..15
//     lanes 16-31: K = k0+16..31
//   B (32x16 f16): lane (l&15) holds column N=n0+(l&15).
//     lanes 0-15 : K = k0+0..15 contiguous in elems 0..15
//     lanes 16-31: K = k0+16..31
//   C/D (16x16 f32): lane half holds 8 rows; N = n0+(l&15).
// ---------------------------------------------------------------------------

__device__ inline v16h load_a_frag_f16(const _Float16* __restrict__ src, int lda,
                                       int row0, int k0) {
  int lane = threadIdx.x & 31;
  int r    = row0 + (lane & 15);
  int kb   = k0 + ((lane & 16) ? 16 : 0);
  const v8h lo = *reinterpret_cast<const v8h*>(src + (size_t)r * lda + kb);
  const v8h hi = *reinterpret_cast<const v8h*>(src + (size_t)r * lda + kb + 8);
  v16h a;
#pragma unroll
  for (int e = 0; e < 8; ++e) { a[e] = lo[e]; a[e + 8] = hi[e]; }
  return a;
}

// A fragment from a wave-private LDS tile laid out [16 rows][32 cols] f16.
__device__ inline v16h load_a_frag_lds(const _Float16* base) {
  int lane = threadIdx.x & 31;
  int r    = lane & 15;
  int kb   = (lane & 16) ? 16 : 0;
  const v8h lo = *reinterpret_cast<const v8h*>(base + r * 32 + kb);
  const v8h hi = *reinterpret_cast<const v8h*>(base + r * 32 + kb + 8);
  v16h a;
#pragma unroll
  for (int e = 0; e < 8; ++e) { a[e] = lo[e]; a[e + 8] = hi[e]; }
  return a;
}

// B fragment where the K-dimension is contiguous in memory per N row:
// Bsrc[n, k] holds B[k, n]  (i.e. we multiply by Bsrc^T). One 32-byte load/lane.
// Used for: Linear weights (y = x @ W^T), K^T in QK^T, V stored transposed.
__device__ inline v16h load_bT_frag_f16(const _Float16* __restrict__ src, int lds_,
                                        int k0, int n0) {
  int lane = threadIdx.x & 31;
  int n    = n0 + (lane & 15);
  int kb   = k0 + ((lane & 16) ? 16 : 0);
  return *reinterpret_cast<const v16h*>(src + (size_t)n * lds_ + kb);
}

__device__ inline v8f wmma_f16(v16h a, v16h b, v8f c) {
  return __builtin_amdgcn_wmma_f32_16x16x32_f16(false, a, false, b, (short)0, c,
                                                false, false);
}

// ---------------------------------------------------------------------------
// f32 -> f16 conversion (grid-stride)
// ---------------------------------------------------------------------------
__global__ void cvt_f32_to_f16(const float* __restrict__ in,
                               _Float16* __restrict__ out, int n) {
  int i = blockIdx.x * blockDim.x + threadIdx.x;
  for (; i < n; i += gridDim.x * blockDim.x) out[i] = (_Float16)in[i];
}

// ---------------------------------------------------------------------------
// QKV projection: Y = x @ W^T, output f16 either in BHLD (mode 0, for Q/K)
// or transposed BHDL (mode 1, for V) so the PV GEMM gets contiguous B loads.
// One wave per 16x16 tile, 8 waves/block. Scale folds 1/sqrt(DH) into Q.
// ---------------------------------------------------------------------------
__global__ void gemm_qkv_kernel(const _Float16* __restrict__ xh,
                                const _Float16* __restrict__ Wh,
                                _Float16* __restrict__ out, float scale,
                                int mode) {
  const int ntn  = kD / 16;                                   // 64 tiles in N
  int tile = blockIdx.x * (blockDim.x >> 5) + (threadIdx.x >> 5);
  int tn = tile % ntn, tm = tile / ntn;
  int row0 = tm * 16, n0 = tn * 16;

  v8f c = {};
  for (int k0 = 0; k0 < kD; k0 += 32) {
    if (k0 + 32 < kD)
      __builtin_prefetch(xh + (size_t)row0 * kD + k0 + 32, 0, 1);
    v16h a = load_a_frag_f16(xh, kD, row0, k0);
    v16h b = load_bT_frag_f16(Wh, kD, k0, n0);                // W[n,k] = B[k,n]
    c = wmma_f16(a, b, c);
  }

  int lane  = threadIdx.x & 31;
  int n     = n0 + (lane & 15);
  int h     = n >> 6;                                         // n / DH
  int dh    = n & (kDH - 1);
  int rbase = row0 + ((lane & 16) ? 8 : 0);
#pragma unroll
  for (int v = 0; v < 8; ++v) {
    int m  = rbase + v;
    int bb = m >> 11;                                         // m / L
    int l  = m & (kL - 1);
    _Float16 val = (_Float16)(c[v] * scale);
    if (mode == 0)       // BHLD: [(b*H+h)*L + l]*DH + dh
      out[((size_t)(bb * kH + h) * kL + l) * kDH + dh] = val;
    else                 // BHDL: [(b*H+h)*DH + dh]*L + l  (transposed V)
      out[((size_t)(bb * kH + h) * kDH + dh) * kL + l] = val;
  }
}

// ---------------------------------------------------------------------------
// Fused attention: per wave, one 16-row strip of one (b,h).
//   Pass 1: stream QK^T tiles through flash-style running (max, sum).
//           Per-lane partials cover one column each; combined across the
//           16-lane halves with __shfl_xor at the end.
//   Pass 2: recompute each score tile (Q/K strip lives in L2: 256 KB/head),
//           normalize, write attn exactly once with NT stores (512 MB
//           streaming output must not rinse the 192 MB L2 holding Q/K/V),
//           stage the probability tile in wave-private LDS to re-layout
//           C-frag -> A-frag, and feed 4 PV WMMAs against transposed V.
//           attn is never re-read.
// ---------------------------------------------------------------------------
__global__ void fused_attn_kernel(const _Float16* __restrict__ Qs,
                                  const _Float16* __restrict__ Kh,
                                  const _Float16* __restrict__ Vt,
                                  float* __restrict__ attn,
                                  _Float16* __restrict__ concat) {
  __shared__ __align__(16) _Float16 stage[8][16][32];         // 8 KB, 1 KB/wave

  int bh = blockIdx.y;
  const _Float16* q  = Qs + (size_t)bh * kL * kDH;
  const _Float16* k  = Kh + (size_t)bh * kL * kDH;
  const _Float16* vt = Vt + (size_t)bh * kDH * kL;            // [dh][m]
  float* s_out       = attn + (size_t)bh * kL * kL;

  int w     = threadIdx.x >> 5;
  int lane  = threadIdx.x & 31;
  int strip = blockIdx.x * 8 + w;                             // 0..127
  int row0  = strip * 16;
  int rbase = row0 + ((lane & 16) ? 8 : 0);
  _Float16* st = &stage[w][0][0];

  // ---- Pass 1: running (max, sum) over all 2048 columns -------------------
  float m[8], ssum[8];
#pragma unroll
  for (int r = 0; r < 8; ++r) { m[r] = -INFINITY; ssum[r] = 0.f; }

  for (int n0 = 0; n0 < kL; n0 += 16) {
    v8f c = {};
#pragma unroll
    for (int k0 = 0; k0 < kDH; k0 += 32) {
      v16h a = load_a_frag_f16(q, kDH, row0, k0);
      v16h b = load_bT_frag_f16(k, kDH, k0, n0);
      c = wmma_f16(a, b, c);
    }
#pragma unroll
    for (int r = 0; r < 8; ++r) {
      float mn = fmaxf(m[r], c[r]);
      ssum[r]  = ssum[r] * __expf(m[r] - mn) + __expf(c[r] - mn);
      m[r]     = mn;
    }
  }
  // Combine per-column partials across each 16-lane half (rows are
  // replicated across those lanes in the C-fragment layout).
#pragma unroll
  for (int xm = 1; xm < 16; xm <<= 1) {
#pragma unroll
    for (int r = 0; r < 8; ++r) {
      float mo = __shfl_xor(m[r], xm, 32);
      float so = __shfl_xor(ssum[r], xm, 32);
      float mn = fmaxf(m[r], mo);
      ssum[r]  = ssum[r] * __expf(m[r] - mn) + so * __expf(mo - mn);
      m[r]     = mn;
    }
  }
  float inv[8];
#pragma unroll
  for (int r = 0; r < 8; ++r) inv[r] = 1.0f / ssum[r];

  // ---- Pass 2: recompute, normalize, emit attn, accumulate PV -------------
  v8f o0 = {}, o1 = {}, o2 = {}, o3 = {};

  for (int g = 0; g < kL; g += 32) {                          // 32-col groups
#pragma unroll
    for (int sub = 0; sub < 32; sub += 16) {
      int n0 = g + sub;
      v8f c = {};
#pragma unroll
      for (int k0 = 0; k0 < kDH; k0 += 32) {
        v16h a = load_a_frag_f16(q, kDH, row0, k0);
        v16h b = load_bT_frag_f16(k, kDH, k0, n0);
        c = wmma_f16(a, b, c);
      }
      int n   = n0 + (lane & 15);
      int col = sub + (lane & 15);
      int rl  = (lane & 16) ? 8 : 0;                          // local row base
#pragma unroll
      for (int r = 0; r < 8; ++r) {
        float p = __expf(c[r] - m[r]) * inv[r];
        // attn is write-once streaming output: non-temporal store.
        __builtin_nontemporal_store(p, s_out + (size_t)(rbase + r) * kL + n);
        st[(rl + r) * 32 + col] = (_Float16)p;                // LDS re-layout
      }
    }
    // Wave-private producer->consumer through LDS: order with DScnt.
    __asm__ volatile("s_wait_dscnt 0" ::: "memory");
    v16h pa = load_a_frag_lds(st);                            // P tile, K=32
    o0 = wmma_f16(pa, load_bT_frag_f16(vt, kL, g, 0),  o0);
    o1 = wmma_f16(pa, load_bT_frag_f16(vt, kL, g, 16), o1);
    o2 = wmma_f16(pa, load_bT_frag_f16(vt, kL, g, 32), o2);
    o3 = wmma_f16(pa, load_bT_frag_f16(vt, kL, g, 48), o3);
  }

  // Store the 16x64 output tile into concat (B, L, H*DH) as f16.
  int bb = bh / kH, h = bh % kH;
  int nloc = lane & 15;
#pragma unroll
  for (int r = 0; r < 8; ++r) {
    int l = rbase + r;
    size_t base = ((size_t)(bb * kL + l)) * kD + h * kDH;
    concat[base + 0  + nloc] = (_Float16)o0[r];
    concat[base + 16 + nloc] = (_Float16)o1[r];
    concat[base + 32 + nloc] = (_Float16)o2[r];
    concat[base + 48 + nloc] = (_Float16)o3[r];
  }
}

// ---------------------------------------------------------------------------
// Output projection: output = concat @ Wo^T, f32 row-major into d_out.
// Final output is never re-read on device -> NT stores.
// ---------------------------------------------------------------------------
__global__ void out_proj_kernel(const _Float16* __restrict__ concat,
                                const _Float16* __restrict__ Woh,
                                float* __restrict__ out) {
  const int ntn = kD / 16;                                    // 64
  int tile = blockIdx.x * (blockDim.x >> 5) + (threadIdx.x >> 5);
  int tn = tile % ntn, tm = tile / ntn;
  int row0 = tm * 16, n0 = tn * 16;

  v8f c = {};
  for (int k0 = 0; k0 < kD; k0 += 32) {
    if (k0 + 32 < kD)
      __builtin_prefetch(concat + (size_t)row0 * kD + k0 + 32, 0, 1);
    v16h a = load_a_frag_f16(concat, kD, row0, k0);
    v16h b = load_bT_frag_f16(Woh, kD, k0, n0);
    c = wmma_f16(a, b, c);
  }

  int lane  = threadIdx.x & 31;
  int n     = n0 + (lane & 15);
  int rbase = row0 + ((lane & 16) ? 8 : 0);
#pragma unroll
  for (int v = 0; v < 8; ++v)
    __builtin_nontemporal_store(c[v], out + (size_t)(rbase + v) * kD + n);
}

// ---------------------------------------------------------------------------
extern "C" void kernel_launch(void* const* d_in, const int* in_sizes, int n_in,
                              void* d_out, int out_size, void* d_ws,
                              size_t ws_size, hipStream_t stream) {
  const float* x  = (const float*)d_in[0];
  const float* Wq = (const float*)d_in[1];
  const float* Wk = (const float*)d_in[2];
  const float* Wv = (const float*)d_in[3];
  const float* Wo = (const float*)d_in[4];

  float* out  = (float*)d_out;                       // (B, L, D) f32
  float* attn = out + (size_t)kB * kL * kD;          // (B, H, L, L) f32

  // Workspace layout (f16 elements): 25,165,824 halves = 48 MB total.
  _Float16* w   = (_Float16*)d_ws;
  _Float16* xh  = w;                                 // 4,194,304
  _Float16* wqh = w + 4194304;                       // 1,048,576
  _Float16* wkh = w + 5242880;
  _Float16* wvh = w + 6291456;
  _Float16* woh = w + 7340032;
  _Float16* qh  = w + 8388608;                       // BHLD, 4,194,304
  _Float16* kh  = w + 12582912;                      // BHLD
  _Float16* vt  = w + 16777216;                      // BHDL (transposed V)
  _Float16* cc  = w + 20971520;                      // concat (B, L, D)

  // 1) f32 -> f16 conversions
  cvt_f32_to_f16<<<2048, 256, 0, stream>>>(x, xh, kM * kD);
  cvt_f32_to_f16<<<512, 256, 0, stream>>>(Wq, wqh, kD * kD);
  cvt_f32_to_f16<<<512, 256, 0, stream>>>(Wk, wkh, kD * kD);
  cvt_f32_to_f16<<<512, 256, 0, stream>>>(Wv, wvh, kD * kD);
  cvt_f32_to_f16<<<512, 256, 0, stream>>>(Wo, woh, kD * kD);

  // 2) QKV projections (16384 tiles, 8 waves/block -> 2048 blocks)
  const float qscale = 0.125f;                       // 1/sqrt(DH)
  gemm_qkv_kernel<<<2048, 256, 0, stream>>>(xh, wqh, qh, qscale, 0);
  gemm_qkv_kernel<<<2048, 256, 0, stream>>>(xh, wkh, kh, 1.0f, 0);
  gemm_qkv_kernel<<<2048, 256, 0, stream>>>(xh, wvh, vt, 1.0f, 1);

  // 3) Fused scores+softmax+PV: 128 strips per (b,h), 8 waves/block.
  fused_attn_kernel<<<dim3(16, kBH), 256, 0, stream>>>(qh, kh, vt, attn, cc);

  // 4) Output projection
  out_proj_kernel<<<2048, 256, 0, stream>>>(cc, woh, out);
}
